// OscillatorySegmenter_23270132810199
// MI455X (gfx1250) — compile-verified, matrix operand order checked by reading, not run on hardware
//
#include <hip/hip_runtime.h>
#include <hip/hip_bf16.h>
#include <math.h>

typedef __attribute__((ext_vector_type(16))) _Float16 v16h;
typedef __attribute__((ext_vector_type(8)))  _Float16 v8h;
typedef __attribute__((ext_vector_type(8)))  float    v8f;
typedef int v4i_gcc __attribute__((vector_size(16)));   // matches builtin param type

#define BB   4
#define NN   1024
#define DD   256
#define HH   4
#define OSC  8
#define DKK  64
#define NUM_STEPS 10
#define DT_F 0.1f
#define THRESHOLD_F 0.5f
#define PI_F 3.14159265358979323846f
#define TWO_PI_F 6.28318530717958647692f

// gfx1250 async global->LDS path (guarded so compile never breaks; the host
// pass and toolchains without the builtin fall back to load+ds_store).
#if defined(__gfx1250__) && __has_builtin(__builtin_amdgcn_global_load_async_to_lds_b128)
#define HAVE_ASYNC_LDS 1
#else
#define HAVE_ASYNC_LDS 0
#endif

// ---------------------------------------------------------------------------
// Fragment loader. ISA 7.12.2 16-bit A (16x32) layout:
//   lane<16  : row M=lane,    K={0..7} in v0..3, K={16..23} in v4..7
//   lane>=16 : row M=lane-16, K={8..15},{24..31}
// B (32x16) mirrors this with "row" = output column (we store B-operands so
// each output column's K-run is a contiguous row of length `ld`).
// ---------------------------------------------------------------------------
__device__ __forceinline__ v16h frag_ld(const _Float16* blockBase, int ld, int kk) {
    const int lane = threadIdx.x & 31;
    const int r    = lane & 15;
    const int sel  = (lane >> 4) << 3;            // 0 or 8
    const _Float16* p = blockBase + (size_t)r * ld + kk + sel;
    v8h lo = *(const v8h*)p;
    v8h hi = *(const v8h*)(p + 16);
    v16h out;
#pragma unroll
    for (int i = 0; i < 8; ++i) { out[i] = lo[i]; out[i + 8] = hi[i]; }
    return out;
}

__device__ __forceinline__ v8f wmma_f16(v16h a, v16h b, v8f c) {
    return __builtin_amdgcn_wmma_f32_16x16x32_f16(
        /*neg_a=*/false, a, /*neg_b=*/false, b,
        /*c_mod=*/(short)0, c, /*reuse_a=*/false, /*reuse_b=*/false);
}

__device__ __forceinline__ float wrap_pi(float x) {
    float t = fmodf(x + PI_F, TWO_PI_F);
    if (t < 0.0f) t += TWO_PI_F;
    return t - PI_F;
}

// ---------------------------------------------------------------------------
__global__ void convert_f32_to_f16(const float* __restrict__ src,
                                   _Float16* __restrict__ dst, int n) {
    int i = blockIdx.x * blockDim.x + threadIdx.x;
    if (i < n) dst[i] = (_Float16)src[i];
}

// ---------------------------------------------------------------------------
// q = x @ Wq^T, k = x @ Wk^T  (f16 WMMA, fp32 accumulate)
// ---------------------------------------------------------------------------
__global__ void proj_qk_kernel(const _Float16* __restrict__ xf,
                               const _Float16* __restrict__ Wqf,
                               const _Float16* __restrict__ Wkf,
                               _Float16* __restrict__ qf,
                               _Float16* __restrict__ kf) {
    const int nColTiles = DD / 16;
    int t  = blockIdx.x;
    int ct = t % nColTiles;
    int rt = t / nColTiles;
    const _Float16* Ab = xf  + (size_t)rt * 16 * DD;
    const _Float16* Bq = Wqf + (size_t)ct * 16 * DD;
    const _Float16* Bk = Wkf + (size_t)ct * 16 * DD;
    v8f accq = {}; v8f acck = {};
#pragma unroll
    for (int kk = 0; kk < DD; kk += 32) {
        v16h a  = frag_ld(Ab, DD, kk);
        v16h bq = frag_ld(Bq, DD, kk);
        v16h bk = frag_ld(Bk, DD, kk);
        accq = wmma_f16(a, bq, accq);
        acck = wmma_f16(a, bk, acck);
    }
    const int lane  = threadIdx.x & 31;
    const int col   = ct * 16 + (lane & 15);
    const int Mbase = (lane >> 4) << 3;
#pragma unroll
    for (int i = 0; i < 8; ++i) {
        size_t row = (size_t)rt * 16 + Mbase + i;
        qf[row * DD + col] = (_Float16)accq[i];
        kf[row * DD + col] = (_Float16)acck[i];
    }
}

// ---------------------------------------------------------------------------
// raw attention scores -> J (f16), scaled by 1/sqrt(DK)=0.125
// ---------------------------------------------------------------------------
__global__ void attn_kernel(const _Float16* __restrict__ qf,
                            const _Float16* __restrict__ kf,
                            _Float16* __restrict__ J) {
    int t     = blockIdx.x;
    int ctile = t % (NN / 16); t /= (NN / 16);
    int rtile = t % (NN / 16); t /= (NN / 16);
    int bh    = t;
    int b = bh >> 2, h = bh & 3;
    const _Float16* Ab = qf + ((size_t)(b * NN + rtile * 16)) * DD + h * DKK;
    const _Float16* Bb = kf + ((size_t)(b * NN + ctile * 16)) * DD + h * DKK;
    v8f acc = {};
#pragma unroll
    for (int kk = 0; kk < DKK; kk += 32) {
        v16h a  = frag_ld(Ab, DD, kk);
        v16h bb = frag_ld(Bb, DD, kk);
        acc = wmma_f16(a, bb, acc);
    }
    const int lane  = threadIdx.x & 31;
    const int col   = ctile * 16 + (lane & 15);
    const int Mbase = (lane >> 4) << 3;
#pragma unroll
    for (int i = 0; i < 8; ++i) {
        size_t row = (size_t)rtile * 16 + Mbase + i;
        J[((size_t)bh * NN + row) * NN + col] = (_Float16)(acc[i] * 0.125f);
    }
}

// ---------------------------------------------------------------------------
// row softmax over J (in place, f16). one wave per row; 32 elems per lane.
// ---------------------------------------------------------------------------
__global__ void softmax_kernel(_Float16* __restrict__ J) {
    _Float16* row = J + (size_t)blockIdx.x * NN;
    const int lane = threadIdx.x & 31;
    float v[32];
    float mx = -3.4e38f;
#pragma unroll
    for (int i = 0; i < 32; ++i) {
        v[i] = (float)row[lane + i * 32];
        mx = fmaxf(mx, v[i]);
    }
#pragma unroll
    for (int off = 16; off >= 1; off >>= 1)
        mx = fmaxf(mx, __shfl_xor(mx, off, 32));
    float s = 0.0f;
#pragma unroll
    for (int i = 0; i < 32; ++i) { v[i] = __expf(v[i] - mx); s += v[i]; }
#pragma unroll
    for (int off = 16; off >= 1; off >>= 1)
        s += __shfl_xor(s, off, 32);
    float inv = 1.0f / s;
#pragma unroll
    for (int i = 0; i < 32; ++i)
        row[lane + i * 32] = (_Float16)(v[i] * inv);
}

// ---------------------------------------------------------------------------
// phases0 = x @ Wp^T + bp ; Pt[b][h][o][n] (f32) and SCt0 (f16, j-major)
// ---------------------------------------------------------------------------
__global__ void phase_init_kernel(const float* __restrict__ x,
                                  const float* __restrict__ Wp,
                                  const float* __restrict__ bp,
                                  float* __restrict__ Pt,
                                  _Float16* __restrict__ SCt0) {
    int idx = blockIdx.x * blockDim.x + threadIdx.x;
    int n = idx & (NN - 1);
    int r = idx >> 10;
    int o = r & (OSC - 1); r >>= 3;
    int h = r & (HH - 1);
    int b = r >> 2;
    const float* xr = x + ((size_t)b * NN + n) * DD;
    const float* wr = Wp + (size_t)(h * OSC + o) * DD;
    float acc = bp[h * OSC + o];
    for (int d = 0; d < DD; ++d) acc = fmaf(xr[d], wr[d], acc);
    Pt[idx] = acc;
    int bh = b * HH + h;
    SCt0[((size_t)(bh * 16 + o)) * NN + n]     = (_Float16)__sinf(acc);
    SCt0[((size_t)(bh * 16 + o + 8)) * NN + n] = (_Float16)__cosf(acc);
}

// ---------------------------------------------------------------------------
// One Kuramoto step. 128-thread blocks (4 waves), one (b,h) + 4 row tiles
// per block. SC slab (16 x N f16) staged into LDS once per block (async
// global->LDS when available), then consumed as WMMA B-fragments via
// ds_load_b128. Four independent accumulator chains hide XDL latency.
// ---------------------------------------------------------------------------
#define SC_LDP (NN + 8)   // +16B row pad: kills 2048B-stride bank conflicts

__global__ void kuramoto_step_kernel(const _Float16* __restrict__ J,
                                     const _Float16* __restrict__ SCt_in,
                                     _Float16* __restrict__ SCt_out,
                                     float* __restrict__ Pt,
                                     const float* __restrict__ omega,
                                     const float* __restrict__ cs_ptr) {
    __shared__ _Float16 scl[16][SC_LDP];
    const int tilesPerBlk = 4;
    const int grpPerBh    = (NN / 16) / tilesPerBlk;       // 16
    const int bh    = blockIdx.x / grpPerBh;
    const int tg    = blockIdx.x % grpPerBh;
    const int wave  = threadIdx.x >> 5;
    const int rtile = tg * tilesPerBlk + wave;
    const int h     = bh & (HH - 1);
    const int n0    = rtile * 16;

    // ---- stage SC (16 x N) for this bh into LDS ----
    {
        const _Float16* src = SCt_in + (size_t)bh * 16 * NN;
        for (int e = threadIdx.x * 8; e < 16 * NN; e += 128 * 8) {
            int row = e >> 10;
            int c   = e & (NN - 1);
            const _Float16* gp = src + (size_t)row * NN + c;
            _Float16* lp = &scl[row][c];
#if HAVE_ASYNC_LDS
            __builtin_amdgcn_global_load_async_to_lds_b128(
                (__attribute__((address_space(1))) v4i_gcc*)(size_t)gp,
                (__attribute__((address_space(3))) v4i_gcc*)(size_t)lp,
                0, 0);
#else
            *(v8h*)lp = *(const v8h*)gp;
#endif
        }
#if HAVE_ASYNC_LDS
#if __has_builtin(__builtin_amdgcn_s_wait_asynccnt)
        __builtin_amdgcn_s_wait_asynccnt(0);
#else
        asm volatile("s_wait_asynccnt 0x0" ::: "memory");
#endif
#endif
        __syncthreads();
    }

    // ---- acc(16x16) = J_tile(16 x N) @ SC(N x 16); 4 accumulator chains ----
    const _Float16* Ab = J + ((size_t)bh * NN + n0) * NN;
    const _Float16* Lb = &scl[0][0];
    v8f acc0 = {}, acc1 = {}, acc2 = {}, acc3 = {};
    for (int mm = 0; mm < NN; mm += 128) {
        v16h a0 = frag_ld(Ab, NN, mm);
        v16h b0 = frag_ld(Lb, SC_LDP, mm);
        acc0 = wmma_f16(a0, b0, acc0);
        v16h a1 = frag_ld(Ab, NN, mm + 32);
        v16h b1 = frag_ld(Lb, SC_LDP, mm + 32);
        acc1 = wmma_f16(a1, b1, acc1);
        v16h a2 = frag_ld(Ab, NN, mm + 64);
        v16h b2 = frag_ld(Lb, SC_LDP, mm + 64);
        acc2 = wmma_f16(a2, b2, acc2);
        v16h a3 = frag_ld(Ab, NN, mm + 96);
        v16h b3 = frag_ld(Lb, SC_LDP, mm + 96);
        acc3 = wmma_f16(a3, b3, acc3);
    }
    v8f acc = (acc0 + acc1) + (acc2 + acc3);

    // ---- pair Js (cols 0..7) with Jc (cols 8..15) across lanes ----
    float part[8];
#pragma unroll
    for (int i = 0; i < 8; ++i) part[i] = __shfl_xor(acc[i], 8, 32);

    const int lane  = threadIdx.x & 31;
    const int j     = lane & 15;
    const int Mbase = (lane >> 4) << 3;
    if (j < 8) {
        v8h s8 = *(const v8h*)&scl[j][n0 + Mbase];
        v8h c8 = *(const v8h*)&scl[j + 8][n0 + Mbase];
        float om = omega[h * OSC + j];
        float cs = cs_ptr[0];
        float* pp = Pt + ((size_t)(bh * OSC + j)) * NN + n0 + Mbase;
        v8h so, co;
#pragma unroll
        for (int i = 0; i < 8; ++i) {
            float Js = acc[i];
            float Jc = part[i];
            float coupling = (float)c8[i] * Js - (float)s8[i] * Jc;
            float ph = pp[i] + DT_F * (om + cs * coupling);
            ph = wrap_pi(ph);
            pp[i] = ph;
            so[i] = (_Float16)__sinf(ph);
            co[i] = (_Float16)__cosf(ph);
        }
        *(v8h*)(SCt_out + ((size_t)(bh * 16 + j)) * NN + n0 + Mbase)     = so;
        *(v8h*)(SCt_out + ((size_t)(bh * 16 + j + 8)) * NN + n0 + Mbase) = co;
    }
}

// ---------------------------------------------------------------------------
__global__ void mean_phase_kernel(const float* __restrict__ Pt,
                                  float* __restrict__ mean_phase) {
    int idx = blockIdx.x * blockDim.x + threadIdx.x;
    int n = idx & (NN - 1);
    int b = idx >> 10;
    float zr = 0.0f, zi = 0.0f;
#pragma unroll
    for (int h = 0; h < HH; ++h)
#pragma unroll
        for (int o = 0; o < OSC; ++o) {
            float ph = Pt[((size_t)((b * HH + h) * OSC + o)) * NN + n];
            zi += __sinf(ph);
            zr += __cosf(ph);
        }
    mean_phase[idx] = atan2f(zi, zr);
}

// ---------------------------------------------------------------------------
__global__ void segment_kernel(const float* __restrict__ mean_phase,
                               int* __restrict__ seg_out,
                               int* __restrict__ nseg_out) {
    __shared__ float mp[NN];
    __shared__ int segA[NN];
    __shared__ int segB[NN];
    __shared__ int cnt;
    const int b = blockIdx.x;
    const int tid = threadIdx.x;
    mp[tid] = mean_phase[b * NN + tid];
    segA[tid] = tid;
    __syncthreads();
    int* cur = segA;
    int* nxt = segB;
    const float a = mp[tid];
    for (int it = 0; it < 12; ++it) {
        int best = NN;
        for (int m = 0; m < NN; ++m) {
            float d = fabsf(a - mp[m]);
            d = fminf(d, TWO_PI_F - d);
            if (d < THRESHOLD_F) best = min(best, cur[m]);
        }
        nxt[tid] = best;
        __syncthreads();
        int* tmp = cur; cur = nxt; nxt = tmp;
    }
    nxt[tid] = 0;
    if (tid == 0) cnt = 0;
    __syncthreads();
    nxt[cur[tid]] = 1;
    __syncthreads();
    atomicAdd(&cnt, nxt[tid]);
    __syncthreads();
    seg_out[b * NN + tid] = cur[tid];
    if (tid == 0) nseg_out[b] = cnt;
}

// ---------------------------------------------------------------------------
extern "C" void kernel_launch(void* const* d_in, const int* in_sizes, int n_in,
                              void* d_out, int out_size, void* d_ws, size_t ws_size,
                              hipStream_t stream) {
    const float* x     = (const float*)d_in[0];
    const float* Wq    = (const float*)d_in[1];
    const float* Wk    = (const float*)d_in[2];
    const float* Wp    = (const float*)d_in[3];
    const float* bp    = (const float*)d_in[4];
    const float* omega = (const float*)d_in[5];
    const float* cs    = (const float*)d_in[6];

    char* ws = (char*)d_ws;
    size_t off = 0;
    auto alloc = [&](size_t bytes) -> char* {
        char* p = ws + off;
        off += (bytes + 255) & ~(size_t)255;
        return p;
    };
    _Float16* xf   = (_Float16*)alloc((size_t)BB * NN * DD * 2);
    _Float16* Wqf  = (_Float16*)alloc((size_t)DD * DD * 2);
    _Float16* Wkf  = (_Float16*)alloc((size_t)DD * DD * 2);
    _Float16* qf   = (_Float16*)alloc((size_t)BB * NN * DD * 2);
    _Float16* kf   = (_Float16*)alloc((size_t)BB * NN * DD * 2);
    _Float16* J    = (_Float16*)alloc((size_t)BB * HH * NN * NN * 2);  // 32 MB, L2-resident
    _Float16* SCt0 = (_Float16*)alloc((size_t)BB * HH * 16 * NN * 2);
    _Float16* SCt1 = (_Float16*)alloc((size_t)BB * HH * 16 * NN * 2);
    float*    Pt   = (float*)   alloc((size_t)BB * HH * OSC * NN * 4);

    float* out_f  = (float*)d_out;
    int*   out_i  = (int*)d_out;
    int*   seg_out   = out_i;
    int*   nseg_out  = out_i + BB * NN;
    float* meanp_out = out_f + BB * NN + BB;

    {
        int n1 = BB * NN * DD;
        convert_f32_to_f16<<<(n1 + 255) / 256, 256, 0, stream>>>(x, xf, n1);
        int n2 = DD * DD;
        convert_f32_to_f16<<<(n2 + 255) / 256, 256, 0, stream>>>(Wq, Wqf, n2);
        convert_f32_to_f16<<<(n2 + 255) / 256, 256, 0, stream>>>(Wk, Wkf, n2);
    }
    proj_qk_kernel<<<(BB * NN / 16) * (DD / 16), 32, 0, stream>>>(xf, Wqf, Wkf, qf, kf);
    attn_kernel<<<BB * HH * (NN / 16) * (NN / 16), 32, 0, stream>>>(qf, kf, J);
    softmax_kernel<<<BB * HH * NN, 32, 0, stream>>>(J);
    phase_init_kernel<<<(BB * HH * OSC * NN) / 256, 256, 0, stream>>>(x, Wp, bp, Pt, SCt0);

    _Float16* scIn = SCt0;
    _Float16* scOut = SCt1;
    for (int s = 0; s < NUM_STEPS; ++s) {
        kuramoto_step_kernel<<<BB * HH * (NN / 16) / 4, 128, 0, stream>>>(
            J, scIn, scOut, Pt, omega, cs);
        _Float16* tmp = scIn; scIn = scOut; scOut = tmp;
    }
    mean_phase_kernel<<<(BB * NN) / 256, 256, 0, stream>>>(Pt, meanp_out);
    segment_kernel<<<BB, NN, 0, stream>>>(meanp_out, seg_out, nseg_out);
}
